// ExpressionEncoder_58265526338285
// MI455X (gfx1250) — matrix operand order repaired and checked by reading, not compiled, not run on hardware
//
#include <hip/hip_runtime.h>

#define BATCH 8
#define DM    256
#define DM2   512
#define NTOK  4096
#define KC    2304      // 256 * 3 * 3
#define BN_EPS 1e-5f

typedef __attribute__((ext_vector_type(16))) __bf16 v16bf;
typedef __attribute__((ext_vector_type(8)))  float  v8f;

union Frag16 { uint4 q[2]; v16bf v; };

__device__ __forceinline__ unsigned short f2bf(float f) {
  unsigned u = __float_as_uint(f);
  u += 0x7FFFu + ((u >> 16) & 1u);          // round-to-nearest-even
  return (unsigned short)(u >> 16);
}

// A-matrix 16x32 bf16 fragment (ISA 7.12.2): lane L holds row (L&15);
// element e -> K = (e&7) + (L>>4)*8 + (e>=8 ? 16 : 0)  => two contiguous 16B runs.
__device__ __forceinline__ v16bf load_afrag(const unsigned short* base, int ld,
                                            int r, int half, int k0) {
  Frag16 f;
  const unsigned short* p = base + (size_t)r * ld + k0 + half * 8;
  f.q[0] = *(const uint4*)p;
  f.q[1] = *(const uint4*)(p + 16);
  return f.v;
}

// B-matrix 32x16 bf16 fragment from a B^T-staged tile [col][K] (stride ld):
// lane L holds column (L&15); element e -> K = (e&7) + (L>>4)*8 + (e>=8 ? 16 : 0) => two 16B runs.
__device__ __forceinline__ v16bf load_bfrag(const unsigned short* base, int ld,
                                            int c, int half, int k0) {
  Frag16 f;
  const unsigned short* p = base + (size_t)c * ld + k0 + half * 8;
  f.q[0] = *(const uint4*)p;
  f.q[1] = *(const uint4*)(p + 16);
  return f.v;
}

__device__ __forceinline__ v8f wmma_bf16(v16bf a, v16bf b, v8f c) {
  return __builtin_amdgcn_wmma_f32_16x16x32_bf16(false, a, false, b,
                                                 (short)0, c, false, false);
}

// CDNA5 async global->LDS copy, 16 bytes per lane, tracked by ASYNCcnt.
// AS(3) pointer value is the byte offset within the workgroup's LDS window,
// which is exactly what the VDST "LDS address VGPR" of
// GLOBAL_LOAD_ASYNC_TO_LDS_B128 wants (GV addressing mode, SADDR=off).
__device__ __forceinline__ void async_copy_b128(const void* gsrc, void* lds_dst) {
  unsigned lds = (unsigned)(unsigned long long)
      (__attribute__((address_space(3))) void*)lds_dst;
  asm volatile("global_load_async_to_lds_b128 %0, %1, off"
               :: "v"(lds), "v"(gsrc) : "memory");
}

__device__ __forceinline__ void wait_asynccnt0() {
  asm volatile("s_wait_asynccnt 0x0" ::: "memory");
}

// ---------------------------------------------------------------- convert
__global__ void k_cvt(const float* __restrict__ s, unsigned short* __restrict__ d, int n) {
  int i = blockIdx.x * blockDim.x + threadIdx.x;
  if (i < n) d[i] = f2bf(s[i]);
}

// ---------------------------------------------------------------- GEMM: q = relu(Wq * X)
// Output token-major qtok[b][n][co]. M = co (A = Wq rows), N = tokens.
__global__ __launch_bounds__(256) void k_gemm_q(
    const unsigned short* __restrict__ xbf, const unsigned short* __restrict__ wqb,
    const float* __restrict__ bq, unsigned short* __restrict__ qtok) {
  __shared__ __align__(16) unsigned short bt[16 * 256];  // B^T: [token][ci]
  const int b  = blockIdx.y;
  const int n0 = blockIdx.x * 16;
  const int t  = threadIdx.x;
  {
    const unsigned short* src = xbf + ((size_t)b * DM + t) * NTOK + n0;
    uint4 a0 = *(const uint4*)src;
    uint4 a1 = *(const uint4*)(src + 8);
    unsigned w[8] = {a0.x, a0.y, a0.z, a0.w, a1.x, a1.y, a1.z, a1.w};
#pragma unroll
    for (int e = 0; e < 16; ++e)
      bt[e * 256 + t] = (unsigned short)(w[e >> 1] >> ((e & 1) * 16));
  }
  __syncthreads();
  const int lane = t & 31, wave = t >> 5;
  const int r = lane & 15, half = lane >> 4;
  const int cb0 = wave * 32;
  v8f acc0 = {}, acc1 = {};
#pragma unroll
  for (int k0 = 0; k0 < DM; k0 += 32) {
    v16bf bfr = load_bfrag(bt, 256, r, half, k0);
    v16bf a0  = load_afrag(wqb + (size_t)cb0 * DM, DM, r, half, k0);
    v16bf a1  = load_afrag(wqb + (size_t)(cb0 + 16) * DM, DM, r, half, k0);
    acc0 = wmma_bf16(a0, bfr, acc0);
    acc1 = wmma_bf16(a1, bfr, acc1);
  }
  const int n = n0 + r;
#pragma unroll
  for (int tile = 0; tile < 2; ++tile) {
    const int cb = cb0 + tile * 16;
    v8f acc = tile ? acc1 : acc0;
    unsigned out[4];
#pragma unroll
    for (int p = 0; p < 4; ++p) {
      float f0 = fmaxf(acc[2 * p]     + bq[cb + half * 8 + 2 * p],     0.f);
      float f1 = fmaxf(acc[2 * p + 1] + bq[cb + half * 8 + 2 * p + 1], 0.f);
      out[p] = (unsigned)f2bf(f0) | ((unsigned)f2bf(f1) << 16);
    }
    unsigned* dst = (unsigned*)(qtok + ((size_t)b * NTOK + n) * DM + cb + half * 8);
    dst[0] = out[0]; dst[1] = out[1]; dst[2] = out[2]; dst[3] = out[3];
  }
}

// ---------------------------------------------------------------- GEMM: kv = relu(Wkv * im2col(X))
// co<256 -> k token-major ktok[b][n][c]; co>=256 -> v channel-major vch[b][c][n].
__global__ __launch_bounds__(256) void k_gemm_kv(
    const unsigned short* __restrict__ xbf, const unsigned short* __restrict__ wkvb,
    const float* __restrict__ bkv, unsigned short* __restrict__ ktok,
    unsigned short* __restrict__ vch) {
  __shared__ __align__(16) unsigned short bt[16 * 32];   // B^T chunk: [token][kk]
  const int b  = blockIdx.y;
  const int n0 = blockIdx.x * 16;
  const int t  = threadIdx.x;
  const int lane = t & 31, wave = t >> 5;
  const int c = lane & 15, half = lane >> 4;
  const int cb0 = wave * 64;
  v8f acc[4] = {};
  for (int k0 = 0; k0 < KC; k0 += 32) {
    __syncthreads();
#pragma unroll
    for (int e = t; e < 512; e += 256) {     // im2col staging
      int nl = e >> 5, kk = e & 31;
      int ki = k0 + kk;
      int ci = ki / 9, rem = ki - ci * 9;
      int kh = rem / 3, kw = rem - kh * 3;
      int n = n0 + nl;
      int h = (n >> 6) + kh - 1, w = (n & 63) + kw - 1;
      unsigned short val = 0;
      if ((unsigned)h < 64u && (unsigned)w < 64u)
        val = xbf[((size_t)b * DM + ci) * NTOK + h * 64 + w];
      bt[nl * 32 + kk] = val;
    }
    __syncthreads();
    v16bf bfr = load_bfrag(bt, 32, c, half, 0);
#pragma unroll
    for (int tile = 0; tile < 4; ++tile) {
      v16bf af = load_afrag(wkvb + (size_t)(cb0 + tile * 16) * KC, KC, c, half, k0);
      acc[tile] = wmma_bf16(af, bfr, acc[tile]);
    }
  }
  const int n = n0 + c;
#pragma unroll
  for (int tile = 0; tile < 4; ++tile) {
    const int cb = cb0 + tile * 16;
    if (cb < DM) {                       // k: token-major, contiguous pack
      unsigned out[4];
#pragma unroll
      for (int p = 0; p < 4; ++p) {
        float f0 = fmaxf(acc[tile][2 * p]     + bkv[cb + half * 8 + 2 * p],     0.f);
        float f1 = fmaxf(acc[tile][2 * p + 1] + bkv[cb + half * 8 + 2 * p + 1], 0.f);
        out[p] = (unsigned)f2bf(f0) | ((unsigned)f2bf(f1) << 16);
      }
      unsigned* dst = (unsigned*)(ktok + ((size_t)b * NTOK + n) * DM + cb + half * 8);
      dst[0] = out[0]; dst[1] = out[1]; dst[2] = out[2]; dst[3] = out[3];
    } else {                             // v: channel-major
      const int cv = cb - DM;
#pragma unroll
      for (int v = 0; v < 8; ++v) {
        float f = fmaxf(acc[tile][v] + bkv[cb + half * 8 + v], 0.f);
        vch[((size_t)b * DM + cv + half * 8 + v) * NTOK + n] = f2bf(f);
      }
    }
  }
}

// ---------------------------------------------------------------- flash attention + residual
// WG = 8 waves, each wave owns 16 query rows; KV processed in 32-wide blocks.
// K/V tiles staged with CDNA5 async global->LDS copies (ASYNCcnt path).
__global__ __launch_bounds__(256) void k_attn(
    const unsigned short* __restrict__ qtok, const unsigned short* __restrict__ ktok,
    const unsigned short* __restrict__ vch, const float* __restrict__ x,
    float* __restrict__ ybuf) {
  __shared__ __align__(16) unsigned short kt[32 * 256];     // [token][d]   (= B^T for QK)
  __shared__ __align__(16) unsigned short vt[256 * 32];     // [d][token]   (= B^T for PV)
  __shared__ __align__(16) unsigned short pl[8 * 16 * 32];  // per-wave P scratch
  const int b = blockIdx.y;
  const int t = threadIdx.x;
  const int lane = t & 31, wave = t >> 5;
  const int r = lane & 15, half = lane >> 4;
  const int q0 = blockIdx.x * 128 + wave * 16;

  v16bf qf[8];
#pragma unroll
  for (int ks = 0; ks < 8; ++ks)
    qf[ks] = load_afrag(qtok + ((size_t)b * NTOK + q0) * DM, DM, r, half, ks * 32);

  v8f zero = {};
  v8f racc[16];
#pragma unroll
  for (int i = 0; i < 16; ++i) racc[i] = zero;
  float mst[8], lst[8];
#pragma unroll
  for (int v = 0; v < 8; ++v) { mst[v] = -3.0e38f; lst[v] = 0.f; }

  const float scale = 0.0625f;   // 1/sqrt(256)
  unsigned short* pw = pl + wave * 512;

  const int krow = t >> 3, kseg = (t & 7) * 32;
  const unsigned short* ksrc0 = ktok + ((size_t)b * NTOK + krow) * DM + kseg;
  const unsigned short* vsrc0 = vch + ((size_t)b * DM + t) * NTOK;

  for (int jb = 0; jb < NTOK / 32; ++jb) {
    const int j0 = jb * 32;
    __syncthreads();
    {   // stage K rows (token-major) -> kt, async direct to LDS
      const unsigned short* src = ksrc0 + (size_t)j0 * DM;
      unsigned short* dst = kt + krow * 256 + kseg;
#pragma unroll
      for (int u = 0; u < 4; ++u)
        async_copy_b128(src + u * 8, dst + u * 8);
    }
    {   // stage V rows (channel-major) -> vt, async direct to LDS
      const unsigned short* src = vsrc0 + j0;
      unsigned short* dst = vt + t * 32;
#pragma unroll
      for (int u = 0; u < 4; ++u)
        async_copy_b128(src + u * 8, dst + u * 8);
    }
    if (jb + 1 < NTOK / 32) {   // prefetch next KV block into L2/WGP$
      __builtin_prefetch(ksrc0 + (size_t)(j0 + 32) * DM, 0, 1);
      __builtin_prefetch(vsrc0 + j0 + 32, 0, 1);
    }
    wait_asynccnt0();
    __syncthreads();
    // S = q . k^T
    v8f s0 = zero, s1 = zero;
#pragma unroll
    for (int ks = 0; ks < 8; ++ks) {
      v16bf b0 = load_bfrag(kt,            256, r, half, ks * 32);
      v16bf b1 = load_bfrag(kt + 16 * 256, 256, r, half, ks * 32);
      s0 = wmma_bf16(qf[ks], b0, s0);
      s1 = wmma_bf16(qf[ks], b1, s1);
    }
    // online softmax (rows live across the 16 lanes of each half-wave)
    float rmax[8], p0[8], p1[8], rsum[8], alpha[8];
#pragma unroll
    for (int v = 0; v < 8; ++v) {
      s0[v] *= scale; s1[v] *= scale;
      rmax[v] = fmaxf(s0[v], s1[v]);
    }
#pragma unroll
    for (int m = 1; m < 16; m <<= 1)
#pragma unroll
      for (int v = 0; v < 8; ++v)
        rmax[v] = fmaxf(rmax[v], __shfl_xor(rmax[v], m, 32));
#pragma unroll
    for (int v = 0; v < 8; ++v) {
      float mn = fmaxf(mst[v], rmax[v]);
      alpha[v] = __expf(mst[v] - mn);
      mst[v] = mn;
      p0[v] = __expf(s0[v] - mn);
      p1[v] = __expf(s1[v] - mn);
      rsum[v] = p0[v] + p1[v];
    }
#pragma unroll
    for (int m = 1; m < 16; m <<= 1)
#pragma unroll
      for (int v = 0; v < 8; ++v)
        rsum[v] += __shfl_xor(rsum[v], m, 32);
#pragma unroll
    for (int v = 0; v < 8; ++v) lst[v] = lst[v] * alpha[v] + rsum[v];
#pragma unroll
    for (int i = 0; i < 16; ++i)
#pragma unroll
      for (int v = 0; v < 8; ++v)
        racc[i][v] *= alpha[v];
    // transpose P (D-layout) -> A-layout via per-wave LDS scratch [16][32]
#pragma unroll
    for (int v = 0; v < 8; ++v) {
      pw[(half * 8 + v) * 32 + r]      = f2bf(p0[v]);
      pw[(half * 8 + v) * 32 + 16 + r] = f2bf(p1[v]);
    }
    __syncthreads();
    v16bf pf = load_afrag(pw, 32, r, half, 0);
    // r += P . V
#pragma unroll
    for (int dt = 0; dt < 16; ++dt) {
      v16bf bf = load_bfrag(vt + dt * 16 * 32, 32, r, half, 0);
      racc[dt] = wmma_bf16(pf, bf, racc[dt]);
    }
  }
  // epilogue: y = r/l + x, channel-major fp32 (for BN)
  float rinv[8];
#pragma unroll
  for (int v = 0; v < 8; ++v) rinv[v] = 1.0f / lst[v];
#pragma unroll
  for (int dt = 0; dt < 16; ++dt) {
    const int d = dt * 16 + r;
    const size_t base = ((size_t)b * DM + d) * NTOK + q0 + half * 8;
    const float4* xr = (const float4*)(x + base);
    float4 o0 = xr[0], o1 = xr[1];
    o0.x += racc[dt][0] * rinv[0]; o0.y += racc[dt][1] * rinv[1];
    o0.z += racc[dt][2] * rinv[2]; o0.w += racc[dt][3] * rinv[3];
    o1.x += racc[dt][4] * rinv[4]; o1.y += racc[dt][5] * rinv[5];
    o1.z += racc[dt][6] * rinv[6]; o1.w += racc[dt][7] * rinv[7];
    float4* yr = (float4*)(ybuf + base);
    yr[0] = o0; yr[1] = o1;
  }
}

// ---------------------------------------------------------------- BatchNorm
__global__ __launch_bounds__(256) void k_bnstats(const float* __restrict__ ybuf,
                                                 float* __restrict__ stats) {
  __shared__ float s1[256], s2[256];
  const int c = blockIdx.x, t = threadIdx.x;
  float sum = 0.f, sq = 0.f;
  for (int i = t; i < BATCH * NTOK; i += 256) {
    int b = i >> 12, n = i & (NTOK - 1);
    float y = ybuf[((size_t)b * DM + c) * NTOK + n];
    sum += y; sq += y * y;
  }
  s1[t] = sum; s2[t] = sq;
  __syncthreads();
  for (int s = 128; s > 0; s >>= 1) {
    if (t < s) { s1[t] += s1[t + s]; s2[t] += s2[t + s]; }
    __syncthreads();
  }
  if (t == 0) {
    float inv = 1.0f / (float)(BATCH * NTOK);
    float mean = s1[0] * inv;
    float var  = s2[0] * inv - mean * mean;
    stats[2 * c]     = mean;
    stats[2 * c + 1] = rsqrtf(var + BN_EPS);
  }
}

__global__ void k_bnapply(const float* __restrict__ ybuf, const float* __restrict__ stats,
                          const float* __restrict__ gamma, const float* __restrict__ beta,
                          float* __restrict__ out, int n) {
  int i = blockIdx.x * blockDim.x + threadIdx.x;
  if (i >= n) return;
  int c = (i >> 12) & (DM - 1);
  float y = ybuf[i];
  out[i] = (y - stats[2 * c]) * stats[2 * c + 1] * gamma[c] + beta[c];
}

// ---------------------------------------------------------------- launch
extern "C" void kernel_launch(void* const* d_in, const int* in_sizes, int n_in,
                              void* d_out, int out_size, void* d_ws, size_t ws_size,
                              hipStream_t stream) {
  const float* x     = (const float*)d_in[0];
  const float* wq    = (const float*)d_in[1];
  const float* bq    = (const float*)d_in[2];
  const float* wkv   = (const float*)d_in[3];
  const float* bkv   = (const float*)d_in[4];
  const float* gamma = (const float*)d_in[5];
  const float* beta  = (const float*)d_in[6];

  char* ws = (char*)d_ws;
  size_t off = 0;
  auto alloc = [&](size_t bytes) {
    char* p = ws + off;
    off += (bytes + 255) & ~(size_t)255;
    return p;
  };
  unsigned short* xbf  = (unsigned short*)alloc((size_t)BATCH * DM * NTOK * 2);
  unsigned short* wqb  = (unsigned short*)alloc((size_t)DM * DM * 2);
  unsigned short* wkvb = (unsigned short*)alloc((size_t)DM2 * KC * 2);
  unsigned short* qtok = (unsigned short*)alloc((size_t)BATCH * NTOK * DM * 2);
  unsigned short* ktok = (unsigned short*)alloc((size_t)BATCH * NTOK * DM * 2);
  unsigned short* vch  = (unsigned short*)alloc((size_t)BATCH * DM * NTOK * 2);
  float* ybuf  = (float*)alloc((size_t)BATCH * DM * NTOK * 4);
  float* stats = (float*)alloc((size_t)DM * 2 * 4);

  const int nx = BATCH * DM * NTOK;
  k_cvt<<<(nx + 255) / 256, 256, 0, stream>>>(x, xbf, nx);
  k_cvt<<<(DM * DM + 255) / 256, 256, 0, stream>>>(wq, wqb, DM * DM);
  k_cvt<<<(DM2 * KC + 255) / 256, 256, 0, stream>>>(wkv, wkvb, DM2 * KC);

  k_gemm_q <<<dim3(NTOK / 16, BATCH), 256, 0, stream>>>(xbf, wqb, bq, qtok);
  k_gemm_kv<<<dim3(NTOK / 16, BATCH), 256, 0, stream>>>(xbf, wkvb, bkv, ktok, vch);
  k_attn   <<<dim3(NTOK / 128, BATCH), 256, 0, stream>>>(qtok, ktok, vch, x, ybuf);

  k_bnstats<<<DM, 256, 0, stream>>>(ybuf, stats);
  k_bnapply<<<(nx + 255) / 256, 256, 0, stream>>>(ybuf, stats, gamma, beta,
                                                  (float*)d_out, nx);
}
